// GAU_4440996184614
// MI455X (gfx1250) — matrix operand order repaired and checked by reading, not compiled, not run on hardware
//
#include <hip/hip_runtime.h>
#include <hip/hip_bf16.h>

#define Bb   32
#define Tt   512
#define Hh   1024
#define Ee   2048
#define Ss   128
#define Ll   512
#define BT   (Bb * Tt)
#define NUV  (2 * Ee + Ss)
#define RSQRT_S 0.08838834764831845f   // 1/sqrt(128)

typedef __attribute__((ext_vector_type(16))) __bf16 v16bf;
typedef __attribute__((ext_vector_type(8)))  float  v8f;
typedef __hip_bfloat16 bf16;

// ---------------------------------------------------------------------------
// WMMA fragment loaders (CDNA5 16-bit layouts, wave32)
// A 16x32 (MxK): lanes 0-15 -> M=lane, K in {0..7}U{16..23}; lanes 16-31 ->
//   M=lane-16, K in {8..15}U{24..31}. Two 16B loads per lane.
// B 32x16 (KxN) from a row-major N x K operand: lane n=lane&15 holds K in
//   {ko..ko+7}U{ko+16..ko+23}, ko=(lane>>4)*8 (same as A). Two 16B loads.
// ---------------------------------------------------------------------------
__device__ __forceinline__ v16bf load_a_frag(const bf16* __restrict__ p, int ld,
                                             int k, int lane) {
    const int row = lane & 15;
    const int ko  = (lane >> 4) << 3;                   // 0 or 8
    const bf16* base = p + (size_t)row * ld + k + ko;
    union { uint4 u[2]; v16bf v; } r;
    r.u[0] = *reinterpret_cast<const uint4*>(base);       // K = ko .. ko+7
    r.u[1] = *reinterpret_cast<const uint4*>(base + 16);  // K = ko+16 .. ko+23
    return r.v;
}

__device__ __forceinline__ v16bf load_b_frag(const bf16* __restrict__ p, int ld,
                                             int k, int lane) {
    const int col = lane & 15;
    const int ko  = (lane >> 4) << 3;                   // 0 or 8
    const bf16* base = p + (size_t)col * ld + k + ko;
    const uint4* q = reinterpret_cast<const uint4*>(base);
    union { uint4 u[2]; v16bf v; } r;
    r.u[0] = q[0];
    r.u[1] = q[2];
    return r.v;
}

// Wave-level GEMM: C(32x64) += A(32xK) * Bt(64xK)^T, 2x4 grid of 16x16 tiles.
// A row-major MxK bf16, Bt row-major NxK bf16. m0/n0 are wave offsets.
// Unconditional speculative prefetch (dropped silently if OOB, no LOADcnt).
__device__ __forceinline__ void gemm_wave(const bf16* __restrict__ A, int lda,
                                          const bf16* __restrict__ Bt, int ldb,
                                          int K, int m0, int n0, v8f acc[2][4]) {
    const int lane = threadIdx.x & 31;
    const bf16* ap = A + (size_t)(m0 + (lane & 15)) * lda;
    const bf16* bp = Bt + (size_t)(n0 + (lane & 15)) * ldb;
#pragma unroll 2
    for (int k = 0; k < K; k += 32) {
        __builtin_prefetch(ap + k + 64, 0, 1);   // global_prefetch_b8
        __builtin_prefetch(bp + k + 64, 0, 1);
        v16bf a[2], b[4];
#pragma unroll
        for (int i = 0; i < 2; ++i)
            a[i] = load_a_frag(A + (size_t)(m0 + i * 16) * lda, lda, k, lane);
#pragma unroll
        for (int j = 0; j < 4; ++j)
            b[j] = load_b_frag(Bt + (size_t)(n0 + j * 16) * ldb, ldb, k, lane);
#pragma unroll
        for (int i = 0; i < 2; ++i)
#pragma unroll
            for (int j = 0; j < 4; ++j)
                acc[i][j] = __builtin_amdgcn_wmma_f32_16x16x32_bf16(
                    false, a[i], false, b[j], (short)0, acc[i][j], false, false);
    }
}

__device__ __forceinline__ void zero_acc(v8f acc[2][4]) {
#pragma unroll
    for (int i = 0; i < 2; ++i)
#pragma unroll
        for (int j = 0; j < 4; ++j) acc[i][j] = (v8f)0.0f;
}

__device__ __forceinline__ float silu_f(float x) {
    return x / (1.0f + __expf(-x));
}

// ---------------------------------------------------------------------------
// 1) fp32 -> bf16 weight conversion
// ---------------------------------------------------------------------------
__global__ void gau_cvt_bf16(const float* __restrict__ in, bf16* __restrict__ out,
                             int n) {
    int i = blockIdx.x * blockDim.x + threadIdx.x;
    if (i < n) out[i] = __float2bfloat16(in[i]);
}

// ---------------------------------------------------------------------------
// 2) LayerNorm over H=1024, write bf16 activations
// ---------------------------------------------------------------------------
__global__ void __launch_bounds__(256) gau_ln(const float* __restrict__ x,
                                              const float* __restrict__ lw,
                                              const float* __restrict__ lb,
                                              bf16* __restrict__ xn) {
    __shared__ float s1[256], s2[256];
    const int row = blockIdx.x;
    const int tid = threadIdx.x;
    const float4 xv = reinterpret_cast<const float4*>(x + (size_t)row * Hh)[tid];
    float s  = xv.x + xv.y + xv.z + xv.w;
    float ss = xv.x * xv.x + xv.y * xv.y + xv.z * xv.z + xv.w * xv.w;
    s1[tid] = s; s2[tid] = ss;
    __syncthreads();
    for (int st = 128; st > 0; st >>= 1) {
        if (tid < st) { s1[tid] += s1[tid + st]; s2[tid] += s2[tid + st]; }
        __syncthreads();
    }
    const float mu  = s1[0] * (1.0f / Hh);
    const float var = s2[0] * (1.0f / Hh) - mu * mu;
    const float inv = rsqrtf(var + 1e-5f);
    const float4 wv = reinterpret_cast<const float4*>(lw)[tid];
    const float4 bv = reinterpret_cast<const float4*>(lb)[tid];
    union { bf16 h[4]; uint2 q; } pk;
    pk.h[0] = __float2bfloat16((xv.x - mu) * inv * wv.x + bv.x);
    pk.h[1] = __float2bfloat16((xv.y - mu) * inv * wv.y + bv.y);
    pk.h[2] = __float2bfloat16((xv.z - mu) * inv * wv.z + bv.z);
    pk.h[3] = __float2bfloat16((xv.w - mu) * inv * wv.w + bv.w);
    reinterpret_cast<uint2*>(xn + (size_t)row * Hh)[tid] = pk.q;
}

// ---------------------------------------------------------------------------
// 3) uv GEMM: silu(xn @ uv_w^T + uv_b) -> u(bf16), v^T(bf16 per batch), base(f32)
// ---------------------------------------------------------------------------
__global__ void __launch_bounds__(128) gau_uv_gemm(const bf16* __restrict__ xn,
                                                   const bf16* __restrict__ uwb,
                                                   const float* __restrict__ uvb,
                                                   bf16* __restrict__ u,
                                                   bf16* __restrict__ vt,
                                                   float* __restrict__ baseo) {
    const int lane = threadIdx.x & 31;
    const int w    = threadIdx.x >> 5;
    const int m0   = blockIdx.y * 64  + (w >> 1) * 32;
    const int n0   = blockIdx.x * 128 + (w & 1) * 64;
    v8f acc[2][4];
    zero_acc(acc);
    gemm_wave(xn, Hh, uwb, Hh, Hh, m0, n0, acc);
#pragma unroll
    for (int i = 0; i < 2; ++i) {
#pragma unroll
        for (int j = 0; j < 4; ++j) {
            const int n  = n0 + j * 16 + (lane & 15);
            const int mb = m0 + i * 16 + ((lane >> 4) << 3);
            const float bias = uvb[n];
            if (n < Ee) {
#pragma unroll
                for (int r = 0; r < 8; ++r)
                    u[(size_t)(mb + r) * Ee + n] =
                        __float2bfloat16(silu_f(acc[i][j][r] + bias));
            } else if (n < 2 * Ee) {
                union { bf16 h[8]; uint4 q; } pk;
#pragma unroll
                for (int r = 0; r < 8; ++r)
                    pk.h[r] = __float2bfloat16(silu_f(acc[i][j][r] + bias));
                const int bb = mb >> 9;          // mb / T
                const int tt = mb & (Tt - 1);    // mb % T
                *reinterpret_cast<uint4*>(vt + (size_t)bb * Ee * Tt +
                                          (size_t)(n - Ee) * Tt + tt) = pk.q;
            } else {
#pragma unroll
                for (int r = 0; r < 8; ++r)
                    baseo[(size_t)(mb + r) * Ss + (n - 2 * Ee)] =
                        silu_f(acc[i][j][r] + bias);
            }
        }
    }
}

// ---------------------------------------------------------------------------
// 4) gamma/beta + RoPE -> q,k bf16.  one block per (b,t), 64 threads
// ---------------------------------------------------------------------------
__global__ void __launch_bounds__(64) gau_rope(const float* __restrict__ baseo,
                                               const float* __restrict__ gamma,
                                               const float* __restrict__ beta,
                                               bf16* __restrict__ qb,
                                               bf16* __restrict__ kb) {
    const int row = blockIdx.x;          // b*T + t
    const int t   = row & (Tt - 1);
    const int i   = threadIdx.x;         // 0..63 (half of S)
    const float inv_freq = __powf(10000.0f, (float)i * (1.0f / 64.0f));
    const float ang = (float)t * inv_freq;
    const float sn = __sinf(ang), cs = __cosf(ang);
    const float* br = baseo + (size_t)row * Ss;
    const float b1 = br[i], b2 = br[i + 64];
    // q (gamma/beta row 0)
    {
        const float x1 = b1 * gamma[i] + beta[i];
        const float x2 = b2 * gamma[i + 64] + beta[i + 64];
        qb[(size_t)row * Ss + i]      = __float2bfloat16(x1 * cs - x2 * sn);
        qb[(size_t)row * Ss + i + 64] = __float2bfloat16(x2 * cs + x1 * sn);
    }
    // k (gamma/beta row 1)
    {
        const float x1 = b1 * gamma[Ss + i] + beta[Ss + i];
        const float x2 = b2 * gamma[Ss + i + 64] + beta[Ss + i + 64];
        kb[(size_t)row * Ss + i]      = __float2bfloat16(x1 * cs - x2 * sn);
        kb[(size_t)row * Ss + i + 64] = __float2bfloat16(x2 * cs + x1 * sn);
    }
}

// ---------------------------------------------------------------------------
// 5) scores = relu(q k^T / sqrt(S) + w[rel])^2   (per batch, bf16 out)
// ---------------------------------------------------------------------------
__global__ void __launch_bounds__(128) gau_scores(const bf16* __restrict__ qb,
                                                  const bf16* __restrict__ kb,
                                                  const float* __restrict__ wrel,
                                                  bf16* __restrict__ scores) {
    const int lane = threadIdx.x & 31;
    const int w    = threadIdx.x >> 5;
    const int bz   = blockIdx.z;
    const int m0   = blockIdx.y * 64  + (w >> 1) * 32;
    const int n0   = blockIdx.x * 128 + (w & 1) * 64;
    v8f acc[2][4];
    zero_acc(acc);
    gemm_wave(qb + (size_t)bz * Tt * Ss, Ss,
              kb + (size_t)bz * Tt * Ss, Ss, Ss, m0, n0, acc);
    bf16* sc = scores + (size_t)bz * Tt * Tt;
#pragma unroll
    for (int i = 0; i < 2; ++i) {
#pragma unroll
        for (int j = 0; j < 4; ++j) {
            const int nj = n0 + j * 16 + (lane & 15);
            const int mb = m0 + i * 16 + ((lane >> 4) << 3);
#pragma unroll
            for (int r = 0; r < 8; ++r) {
                const int mi = mb + r;
                float v = acc[i][j][r] * RSQRT_S + wrel[nj - mi + (Ll - 1)];
                float rl = fmaxf(v, 0.0f);
                sc[(size_t)mi * Tt + nj] = __float2bfloat16(rl * rl);
            }
        }
    }
}

// ---------------------------------------------------------------------------
// 6) o = u * (scores @ v)   (per batch; v^T layout makes it an N x K operand)
// ---------------------------------------------------------------------------
__global__ void __launch_bounds__(128) gau_av(const bf16* __restrict__ scores,
                                              const bf16* __restrict__ vt,
                                              const bf16* __restrict__ u,
                                              bf16* __restrict__ o) {
    const int lane = threadIdx.x & 31;
    const int w    = threadIdx.x >> 5;
    const int bz   = blockIdx.z;
    const int m0   = blockIdx.y * 64  + (w >> 1) * 32;
    const int n0   = blockIdx.x * 128 + (w & 1) * 64;
    v8f acc[2][4];
    zero_acc(acc);
    gemm_wave(scores + (size_t)bz * Tt * Tt, Tt,
              vt + (size_t)bz * Ee * Tt, Tt, Tt, m0, n0, acc);
#pragma unroll
    for (int i = 0; i < 2; ++i) {
#pragma unroll
        for (int j = 0; j < 4; ++j) {
            const int nj = n0 + j * 16 + (lane & 15);
            const int mb = m0 + i * 16 + ((lane >> 4) << 3);
#pragma unroll
            for (int r = 0; r < 8; ++r) {
                const size_t idx = ((size_t)bz * Tt + mb + r) * Ee + nj;
                o[idx] = __float2bfloat16(acc[i][j][r] * __bfloat162float(u[idx]));
            }
        }
    }
}

// ---------------------------------------------------------------------------
// 7) out = o @ o_w^T + o_b + x  (fp32 output)
// ---------------------------------------------------------------------------
__global__ void __launch_bounds__(128) gau_final(const bf16* __restrict__ o,
                                                 const bf16* __restrict__ owb,
                                                 const float* __restrict__ ob,
                                                 const float* __restrict__ x,
                                                 float* __restrict__ out) {
    const int lane = threadIdx.x & 31;
    const int w    = threadIdx.x >> 5;
    const int m0   = blockIdx.y * 64  + (w >> 1) * 32;
    const int n0   = blockIdx.x * 128 + (w & 1) * 64;
    v8f acc[2][4];
    zero_acc(acc);
    gemm_wave(o, Ee, owb, Ee, Ee, m0, n0, acc);
#pragma unroll
    for (int i = 0; i < 2; ++i) {
#pragma unroll
        for (int j = 0; j < 4; ++j) {
            const int nj = n0 + j * 16 + (lane & 15);
            const int mb = m0 + i * 16 + ((lane >> 4) << 3);
            const float bias = ob[nj];
#pragma unroll
            for (int r = 0; r < 8; ++r) {
                const size_t idx = (size_t)(mb + r) * Hh + nj;
                out[idx] = acc[i][j][r] + bias + x[idx];
            }
        }
    }
}

// ---------------------------------------------------------------------------
// host launcher
// ---------------------------------------------------------------------------
extern "C" void kernel_launch(void* const* d_in, const int* in_sizes, int n_in,
                              void* d_out, int out_size, void* d_ws, size_t ws_size,
                              hipStream_t stream) {
    const float* x     = (const float*)d_in[0];
    const float* ln_w  = (const float*)d_in[1];
    const float* ln_b  = (const float*)d_in[2];
    const float* uv_w  = (const float*)d_in[3];
    const float* uv_b  = (const float*)d_in[4];
    const float* gamma = (const float*)d_in[5];
    const float* beta  = (const float*)d_in[6];
    const float* wrel  = (const float*)d_in[7];
    const float* o_w   = (const float*)d_in[8];
    const float* o_b   = (const float*)d_in[9];
    float* out = (float*)d_out;

    char* ws = (char*)d_ws;
    size_t off = 0;
    auto alloc = [&](size_t bytes) -> char* {
        char* p = ws + off;
        off += (bytes + 255) & ~(size_t)255;
        return p;
    };
    bf16*  xn    = (bf16*)alloc((size_t)BT * Hh * 2);
    bf16*  uwb   = (bf16*)alloc((size_t)NUV * Hh * 2);
    bf16*  owb   = (bf16*)alloc((size_t)Hh * Ee * 2);
    bf16*  u     = (bf16*)alloc((size_t)BT * Ee * 2);
    bf16*  vt    = (bf16*)alloc((size_t)Bb * Ee * Tt * 2);
    float* baseo = (float*)alloc((size_t)BT * Ss * 4);
    bf16*  qb    = (bf16*)alloc((size_t)BT * Ss * 2);
    bf16*  kb    = (bf16*)alloc((size_t)BT * Ss * 2);
    bf16*  sc    = (bf16*)alloc((size_t)Bb * Tt * Tt * 2);
    bf16*  o     = (bf16*)alloc((size_t)BT * Ee * 2);

    // 1) weight conversion
    {
        int n1 = NUV * Hh;
        gau_cvt_bf16<<<(n1 + 255) / 256, 256, 0, stream>>>(uv_w, uwb, n1);
        int n2 = Hh * Ee;
        gau_cvt_bf16<<<(n2 + 255) / 256, 256, 0, stream>>>(o_w, owb, n2);
    }
    // 2) layernorm
    gau_ln<<<BT, 256, 0, stream>>>(x, ln_w, ln_b, xn);
    // 3) uv projection
    gau_uv_gemm<<<dim3(NUV / 128, BT / 64), 128, 0, stream>>>(xn, uwb, uv_b,
                                                              u, vt, baseo);
    // 4) rope
    gau_rope<<<BT, 64, 0, stream>>>(baseo, gamma, beta, qb, kb);
    // 5) attention scores
    gau_scores<<<dim3(Tt / 128, Tt / 64, Bb), 128, 0, stream>>>(qb, kb, wrel, sc);
    // 6) scores @ v, gated by u
    gau_av<<<dim3(Ee / 128, Tt / 64, Bb), 128, 0, stream>>>(sc, vt, u, o);
    // 7) output projection + residual
    gau_final<<<dim3(Hh / 128, BT / 64), 128, 0, stream>>>(o, owb, o_b, x, out);
}
